// MultiHeadAttention_21715354649218
// MI455X (gfx1250) — compile-verified, matrix operand order checked by reading, not run on hardware
//
#include <hip/hip_runtime.h>
#include <hip/hip_bf16.h>

// MHA forward for MI455X (gfx1250, wave32, WMMA, async global->LDS).
// B=4, T=2048, E=1024, H=16, D=64.  Workspace use: 64 MB.

#define T_SEQ 2048
#define EMB   1024
#define NHEAD 16
#define HDIM  64
#define BT    8192          // 4 * 2048
#define QKV_ELEMS ((size_t)4 * NHEAD * T_SEQ * HDIM)   // 8,388,608 bf16 each

typedef __attribute__((ext_vector_type(16))) __bf16        v16bf;
typedef __attribute__((ext_vector_type(8)))  float         v8f;
typedef __attribute__((ext_vector_type(16))) unsigned short ush16;
typedef __attribute__((ext_vector_type(8)))  unsigned short ush8;
typedef __attribute__((ext_vector_type(4)))  unsigned short ush4;

static __device__ __forceinline__ unsigned short f2bfu(float f) {
  unsigned int u = __float_as_uint(f);
  u += 0x7FFFu + ((u >> 16) & 1u);          // round-to-nearest-even
  return (unsigned short)(u >> 16);
}

union U16 { ush16 u; v16bf b; };

// Assemble a 16-element bf16 A/B fragment for v_wmma_*_16x16x32 from a
// row-major 32-wide k-run.  Per ISA 7.12.2: element e maps to
// k = 16*(e>>3) + 8*(lane>>4) + (e&7)  -> two contiguous 8-short (16B) loads.
static __device__ __forceinline__ v16bf make_frag(const unsigned short* rowp, int half) {
  ush8 lo = *(const ush8*)(rowp + half * 8);
  ush8 hi = *(const ush8*)(rowp + 16 + half * 8);
  U16 r;
#pragma unroll
  for (int i = 0; i < 8; ++i) { r.u[i] = lo[i]; r.u[i + 8] = hi[i]; }
  return r.b;
}

static __device__ __forceinline__ v8f wmma_bf16(v16bf a, v16bf b, v8f c) {
  return __builtin_amdgcn_wmma_f32_16x16x32_bf16(false, a, false, b, (short)0, c,
                                                 false, false);
}

// Async copy 16B global -> LDS, tracked on ASYNCcnt (CDNA5 GLOBAL_LOAD_ASYNC).
// LDS address = low 32 bits of the generic shared pointer (ISA 10.2 aperture).
static __device__ __forceinline__ void async_copy_b128(const unsigned short* gsrc,
                                                       unsigned short* ldst) {
  unsigned int laddr = (unsigned int)(size_t)ldst;
  unsigned long long gaddr = (unsigned long long)(size_t)gsrc;
  asm volatile("global_load_async_to_lds_b128 %0, %1, off"
               :: "v"(laddr), "v"(gaddr) : "memory");
}
static __device__ __forceinline__ void wait_async0() {
  asm volatile("s_wait_asynccnt 0x0" ::: "memory");
}

static __device__ __forceinline__ float rmax16(float x) {
  x = fmaxf(x, __shfl_xor(x, 1, 32));
  x = fmaxf(x, __shfl_xor(x, 2, 32));
  x = fmaxf(x, __shfl_xor(x, 4, 32));
  x = fmaxf(x, __shfl_xor(x, 8, 32));
  return x;
}
static __device__ __forceinline__ float rsum16(float x) {
  x += __shfl_xor(x, 1, 32);
  x += __shfl_xor(x, 2, 32);
  x += __shfl_xor(x, 4, 32);
  x += __shfl_xor(x, 8, 32);
  return x;
}

// ---------------------------------------------------------------------------
// Kernel 1: fused QKV projection.  y = x @ W.T + b (Q also * 0.125),
// output bf16 in [B, H, T, D] layout.  Tile: 128 (tokens) x 64 (one head),
// k-chunk 64 per barrier round (8 WMMA / round).
// ---------------------------------------------------------------------------
__global__ __launch_bounds__(256)
void mha_qkv_kernel(const float* __restrict__ x,
                    const float* __restrict__ Wq, const float* __restrict__ bq,
                    const float* __restrict__ Wk, const float* __restrict__ bk,
                    const float* __restrict__ Wv, const float* __restrict__ bv,
                    unsigned short* __restrict__ qout,
                    unsigned short* __restrict__ kout,
                    unsigned short* __restrict__ vout) {
  __shared__ unsigned short aLds[128 * 72];   // 128 rows x 64 k (pad to 72)
  __shared__ unsigned short bLds[64 * 72];    // 64 out-cols x 64 k

  const int z = blockIdx.z;
  const float* W    = (z == 0) ? Wq : ((z == 1) ? Wk : Wv);
  const float* bias = (z == 0) ? bq : ((z == 1) ? bk : bv);
  unsigned short* dst = (z == 0) ? qout : ((z == 1) ? kout : vout);
  const float scale = (z == 0) ? 0.125f : 1.0f;   // HEAD_DIM^-0.5 = 64^-0.5

  const int tid  = threadIdx.x;
  const int wave = tid >> 5, lane = tid & 31;
  const int lid  = lane & 15, half = lane >> 4;
  const int wm = wave >> 1, wn = wave & 1;        // 4x2 wave grid
  const int rowBase = blockIdx.x * 128;
  const int nBase   = blockIdx.y * 64;

  const v8f vzero = {0.f, 0.f, 0.f, 0.f, 0.f, 0.f, 0.f, 0.f};
  v8f acc[2][2];
#pragma unroll
  for (int i = 0; i < 2; ++i)
#pragma unroll
    for (int j = 0; j < 2; ++j) acc[i][j] = vzero;

  for (int kb = 0; kb < EMB / 64; ++kb) {
    __syncthreads();
    // stage A tile (f32 -> bf16): 128x64
#pragma unroll
    for (int p = 0; p < 8; ++p) {
      int r = p * 16 + (tid >> 4), c4 = tid & 15;
      float4 f = *(const float4*)&x[(size_t)(rowBase + r) * EMB + kb * 64 + c4 * 4];
      ush4 s; s[0] = f2bfu(f.x); s[1] = f2bfu(f.y); s[2] = f2bfu(f.z); s[3] = f2bfu(f.w);
      *(ush4*)&aLds[r * 72 + c4 * 4] = s;
    }
    // stage B tile: W[n][k], 64x64
#pragma unroll
    for (int p = 0; p < 4; ++p) {
      int n = p * 16 + (tid >> 4), c4 = tid & 15;
      float4 f = *(const float4*)&W[(size_t)(nBase + n) * EMB + kb * 64 + c4 * 4];
      ush4 s; s[0] = f2bfu(f.x); s[1] = f2bfu(f.y); s[2] = f2bfu(f.z); s[3] = f2bfu(f.w);
      *(ush4*)&bLds[n * 72 + c4 * 4] = s;
    }
    __syncthreads();

#pragma unroll
    for (int g = 0; g < 2; ++g) {
      v16bf a0 = make_frag(&aLds[(wm * 32 + lid) * 72 + g * 32], half);
      v16bf a1 = make_frag(&aLds[(wm * 32 + 16 + lid) * 72 + g * 32], half);
      v16bf b0 = make_frag(&bLds[(wn * 32 + lid) * 72 + g * 32], half);
      v16bf b1 = make_frag(&bLds[(wn * 32 + 16 + lid) * 72 + g * 32], half);
      acc[0][0] = wmma_bf16(a0, b0, acc[0][0]);
      acc[0][1] = wmma_bf16(a0, b1, acc[0][1]);
      acc[1][0] = wmma_bf16(a1, b0, acc[1][0]);
      acc[1][1] = wmma_bf16(a1, b1, acc[1][1]);
    }
  }

  // epilogue: +bias, *scale, bf16 store scattered into [B,H,T,D]
#pragma unroll
  for (int mi = 0; mi < 2; ++mi)
#pragma unroll
    for (int ni = 0; ni < 2; ++ni) {
      int colG = nBase + wn * 32 + ni * 16 + lid;
      float bb = bias[colG];
      int h = colG >> 6, d = colG & 63;
#pragma unroll
      for (int v = 0; v < 8; ++v) {
        int rowG = rowBase + wm * 32 + mi * 16 + half * 8 + v;
        int b = rowG >> 11, t = rowG & (T_SEQ - 1);
        float val = (acc[mi][ni][v] + bb) * scale;
        dst[((size_t)(b * NHEAD + h) * T_SEQ + t) * HDIM + d] = f2bfu(val);
      }
    }
}

// ---------------------------------------------------------------------------
// Kernel 2: flash attention.  One block = 128 query rows of one (b,h).
// 8 waves; wave w owns query rows [w*16, w*16+16).  32 keys per step.
// K tile staged via GLOBAL_LOAD_ASYNC_TO_LDS (ASYNCcnt); V transposed in LDS.
// ---------------------------------------------------------------------------
__global__ __launch_bounds__(256)
void mha_flash_kernel(const unsigned short* __restrict__ q,
                      const unsigned short* __restrict__ k,
                      const unsigned short* __restrict__ vglob,
                      unsigned short* __restrict__ attn_out) {
  __shared__ unsigned short Klds[32 * 72];        // [key][d], pad 64->72
  __shared__ unsigned short Vt[64 * 40];          // [d][key], pad 32->40
  __shared__ unsigned short pLds[8 * 16 * 40];    // per-wave P staging

  const int tid  = threadIdx.x;
  const int wave = tid >> 5, lane = tid & 31;
  const int lid  = lane & 15, half = lane >> 4;
  const int bh = blockIdx.y;                       // b*16 + h
  const int b  = bh >> 4, h = bh & 15;
  const int qBase = blockIdx.x * 128;

  // Q fragments for this wave's 16 rows (already scaled by 0.125 in kernel 1)
  const unsigned short* qrowp =
      q + ((size_t)bh * T_SEQ + qBase + wave * 16 + lid) * HDIM;
  v16bf qa0 = make_frag(qrowp, half);        // d = 0..31
  v16bf qa1 = make_frag(qrowp + 32, half);   // d = 32..63

  const v8f vzero = {0.f, 0.f, 0.f, 0.f, 0.f, 0.f, 0.f, 0.f};
  v8f o[4];                                  // 16 x 64 f32 accumulator
#pragma unroll
  for (int nt = 0; nt < 4; ++nt) o[nt] = vzero;
  float mrun[8], lrun[8];
#pragma unroll
  for (int v = 0; v < 8; ++v) { mrun[v] = -1e30f; lrun[v] = 0.f; }

  unsigned short* pw = &pLds[wave * 16 * 40];
  const int key = tid >> 3, d8 = tid & 7;    // K/V tile staging split

  for (int kb = 0; kb < T_SEQ / 32; ++kb) {
    __syncthreads();
    {
      size_t gofs = ((size_t)bh * T_SEQ + kb * 32 + key) * HDIM + d8 * 8;
      // K tile: async DMA straight into LDS (no VGPR round-trip)
      async_copy_b128(&k[gofs], &Klds[key * 72 + d8 * 8]);
      // V tile: load + transpose-scatter into Vt[d][key]
      ush8 vv = *(const ush8*)&vglob[gofs];
#pragma unroll
      for (int i = 0; i < 8; ++i) Vt[(d8 * 8 + i) * 40 + key] = vv[i];
      if (kb + 1 < T_SEQ / 32)               // prefetch next V tile
        __builtin_prefetch(&vglob[gofs + 32 * HDIM], 0, 1);
    }
    wait_async0();                           // K landed in LDS
    __syncthreads();

    // S = Q * K^T : two 16x16 score tiles per wave, k-dim = 64 = 2 steps
    v8f s0 = vzero, s1 = vzero;
#pragma unroll
    for (int g = 0; g < 2; ++g) {
      v16bf bk0 = make_frag(&Klds[lid * 72 + g * 32], half);
      v16bf bk1 = make_frag(&Klds[(16 + lid) * 72 + g * 32], half);
      v16bf aq = (g == 0) ? qa0 : qa1;
      s0 = wmma_bf16(aq, bk0, s0);
      s1 = wmma_bf16(aq, bk1, s1);
    }

    // online softmax over the 32 columns (rows span 16 lanes of a half-wave)
    float corr[8];
#pragma unroll
    for (int v = 0; v < 8; ++v) {
      float mx = rmax16(fmaxf(s0[v], s1[v]));
      float mnew = fmaxf(mrun[v], mx);
      float p0 = __expf(s0[v] - mnew);
      float p1 = __expf(s1[v] - mnew);
      s0[v] = p0; s1[v] = p1;
      float rs = rsum16(p0 + p1);
      corr[v] = __expf(mrun[v] - mnew);
      lrun[v] = lrun[v] * corr[v] + rs;
      mrun[v] = mnew;
    }
#pragma unroll
    for (int nt = 0; nt < 4; ++nt)
#pragma unroll
      for (int v = 0; v < 8; ++v) o[nt][v] *= corr[v];

    // C-layout -> A-layout transpose of P through wave-private LDS
#pragma unroll
    for (int v = 0; v < 8; ++v) {
      int row = half * 8 + v;
      pw[row * 40 + lid]      = f2bfu(s0[v]);
      pw[row * 40 + 16 + lid] = f2bfu(s1[v]);
    }
    v16bf pa = make_frag(&pw[lid * 40], half);     // LDS in-order within wave

    // O += P * V  (V transposed in LDS: Vt[d][key])
#pragma unroll
    for (int nt = 0; nt < 4; ++nt) {
      v16bf bvf = make_frag(&Vt[(nt * 16 + lid) * 40], half);
      o[nt] = wmma_bf16(pa, bvf, o[nt]);
    }
  }

  // normalize and store bf16 into [B*T, E] for the output projection
  float inv[8];
#pragma unroll
  for (int v = 0; v < 8; ++v) inv[v] = 1.0f / lrun[v];
#pragma unroll
  for (int nt = 0; nt < 4; ++nt)
#pragma unroll
    for (int v = 0; v < 8; ++v) {
      int t = qBase + wave * 16 + half * 8 + v;
      int col = h * HDIM + nt * 16 + lid;
      attn_out[(size_t)(b * T_SEQ + t) * EMB + col] = f2bfu(o[nt][v] * inv[v]);
    }
}

// ---------------------------------------------------------------------------
// Kernel 3: output projection.  out = attn @ Wo.T + bo, f32 result.
// A tile (already bf16) staged via async global->LDS; k-chunk 64.
// ---------------------------------------------------------------------------
__global__ __launch_bounds__(256)
void mha_oproj_kernel(const unsigned short* __restrict__ a,
                      const float* __restrict__ Wo,
                      const float* __restrict__ bo,
                      float* __restrict__ out) {
  __shared__ unsigned short aLds[128 * 72];
  __shared__ unsigned short bLds[64 * 72];

  const int tid  = threadIdx.x;
  const int wave = tid >> 5, lane = tid & 31;
  const int lid  = lane & 15, half = lane >> 4;
  const int wm = wave >> 1, wn = wave & 1;
  const int rowBase = blockIdx.x * 128;
  const int nBase   = blockIdx.y * 64;

  const v8f vzero = {0.f, 0.f, 0.f, 0.f, 0.f, 0.f, 0.f, 0.f};
  v8f acc[2][2];
#pragma unroll
  for (int i = 0; i < 2; ++i)
#pragma unroll
    for (int j = 0; j < 2; ++j) acc[i][j] = vzero;

  for (int kb = 0; kb < EMB / 64; ++kb) {
    __syncthreads();
    // A tile: bf16, 128x64, async DMA into LDS (16B per lane per op)
#pragma unroll
    for (int p = 0; p < 4; ++p) {
      int idx = p * 256 + tid;
      int r = idx >> 3, c8 = idx & 7;
      async_copy_b128(&a[(size_t)(rowBase + r) * EMB + kb * 64 + c8 * 8],
                      &aLds[r * 72 + c8 * 8]);
    }
    // B tile: Wo f32 -> bf16 (must pass through VGPRs for the convert)
#pragma unroll
    for (int p = 0; p < 4; ++p) {
      int n = p * 16 + (tid >> 4), c4 = tid & 15;
      float4 f = *(const float4*)&Wo[(size_t)(nBase + n) * EMB + kb * 64 + c4 * 4];
      ush4 s; s[0] = f2bfu(f.x); s[1] = f2bfu(f.y); s[2] = f2bfu(f.z); s[3] = f2bfu(f.w);
      *(ush4*)&bLds[n * 72 + c4 * 4] = s;
    }
    wait_async0();
    __syncthreads();

#pragma unroll
    for (int g = 0; g < 2; ++g) {
      v16bf a0 = make_frag(&aLds[(wm * 32 + lid) * 72 + g * 32], half);
      v16bf a1 = make_frag(&aLds[(wm * 32 + 16 + lid) * 72 + g * 32], half);
      v16bf b0 = make_frag(&bLds[(wn * 32 + lid) * 72 + g * 32], half);
      v16bf b1 = make_frag(&bLds[(wn * 32 + 16 + lid) * 72 + g * 32], half);
      acc[0][0] = wmma_bf16(a0, b0, acc[0][0]);
      acc[0][1] = wmma_bf16(a0, b1, acc[0][1]);
      acc[1][0] = wmma_bf16(a1, b0, acc[1][0]);
      acc[1][1] = wmma_bf16(a1, b1, acc[1][1]);
    }
  }

#pragma unroll
  for (int mi = 0; mi < 2; ++mi)
#pragma unroll
    for (int ni = 0; ni < 2; ++ni) {
      int colG = nBase + wn * 32 + ni * 16 + lid;
      float bb = bo[colG];
#pragma unroll
      for (int v = 0; v < 8; ++v) {
        int rowG = rowBase + wm * 32 + mi * 16 + half * 8 + v;
        out[(size_t)rowG * EMB + colG] = acc[mi][ni][v] + bb;
      }
    }
}

// ---------------------------------------------------------------------------
extern "C" void kernel_launch(void* const* d_in, const int* in_sizes, int n_in,
                              void* d_out, int out_size, void* d_ws, size_t ws_size,
                              hipStream_t stream) {
  const float* x  = (const float*)d_in[0];
  const float* Wq = (const float*)d_in[1];
  const float* bq = (const float*)d_in[2];
  const float* Wk = (const float*)d_in[3];
  const float* bk = (const float*)d_in[4];
  const float* Wv = (const float*)d_in[5];
  const float* bv = (const float*)d_in[6];
  const float* Wo = (const float*)d_in[7];
  const float* bo = (const float*)d_in[8];
  float* out = (float*)d_out;

  // workspace layout (bf16): Q | K | V (each [B,H,T,D]) | attn ([B*T, E])
  unsigned short* q_bf    = (unsigned short*)d_ws;
  unsigned short* k_bf    = q_bf + QKV_ELEMS;
  unsigned short* v_bf    = k_bf + QKV_ELEMS;
  unsigned short* attn_bf = v_bf + QKV_ELEMS;

  dim3 gQKV(BT / 128, EMB / 64, 3);
  mha_qkv_kernel<<<gQKV, 256, 0, stream>>>(x, Wq, bq, Wk, bk, Wv, bv,
                                           q_bf, k_bf, v_bf);

  dim3 gAtt(T_SEQ / 128, 4 * NHEAD);
  mha_flash_kernel<<<gAtt, 256, 0, stream>>>(q_bf, k_bf, v_bf, attn_bf);

  dim3 gO(BT / 128, EMB / 64);
  mha_oproj_kernel<<<gO, 256, 0, stream>>>(attn_bf, Wo, bo, out);
}
